// LSTM2d_27152783245356
// MI455X (gfx1250) — matrix operand order repaired and checked by reading, not compiled, not run on hardware
//
#include <hip/hip_runtime.h>

typedef unsigned short u16;
typedef __attribute__((ext_vector_type(16))) __bf16 v16bf;
typedef __attribute__((ext_vector_type(8)))  float  v8f;
typedef __attribute__((ext_vector_type(8)))  u16    v8u;
typedef __attribute__((ext_vector_type(16))) u16    v16u;

#define BN 64
#define IN 48
#define ON 48
#define FN 256
#define HN 256
#define NG 1280   // 5*HN
#define KW 512    // 2*FN == 2*HN
#define LDP 520   // padded LDS row stride in u16 (512 + 8 -> 4-bank rotation/row)

__device__ __forceinline__ u16 f2bf(float f) {
  unsigned u = __builtin_bit_cast(unsigned, f);
  u += 0x7FFFu + ((u >> 16) & 1u);   // round-to-nearest-even
  return (u16)(u >> 16);
}

union FragA { v8u h[2]; v16bf v; };
union FragB { v16u h; v16bf v; };

// A-matrix 16x32 bf16 fragment (CDNA5 ISA layout):
// lane<16: M=lane,   K = {k0..k0+7, k0+16..k0+23}
// lane>=16: M=lane-16, K = {k0+8..k0+15, k0+24..k0+31}
__device__ __forceinline__ v16bf load_a(const u16* __restrict__ base, int ldk,
                                        int m0, int k0, int lane) {
  int half = lane >> 4, m = lane & 15;
  const u16* p = base + (size_t)(m0 + m) * ldk + k0 + half * 8;
  FragA u;
  u.h[0] = *(const v8u*)p;
  u.h[1] = *(const v8u*)(p + 16);
  return u.v;
}

// B-matrix 32x16 bf16 fragment from transposed weights WT[n][k] in global:
// lane<16: N=lane, K={k0..k0+7, k0+16..k0+23} ; lane>=16: N=lane-16, K={k0+8..k0+15, k0+24..k0+31}
__device__ __forceinline__ v16bf load_b(const u16* __restrict__ baseT, int ldk,
                                        int n0, int k0, int lane) {
  int half = lane >> 4, n = lane & 15;
  const u16* p = baseT + (size_t)(n0 + n) * ldk + k0 + half * 8;
  FragA u;
  u.h[0] = *(const v8u*)p;
  u.h[1] = *(const v8u*)(p + 16);
  return u.v;
}

// B fragment from LDS-staged tile (two ds_load_b128)
__device__ __forceinline__ v16bf load_b_lds(const u16* p) {
  FragA u;
  u.h[0] = *(const v8u*)p;
  u.h[1] = *(const v8u*)(p + 16);
  return u.v;
}

// (B,I,F) f32 -> [i][b][f] bf16   (works for x and y since I==O)
__global__ void k_cvt_act(const float* __restrict__ src, u16* __restrict__ dst) {
  int t = blockIdx.x * blockDim.x + threadIdx.x;
  int f  = t & (FN - 1);
  int ib = t >> 8;
  int b  = ib & (BN - 1);
  int i  = ib >> 6;
  dst[t] = f2bf(src[(size_t)(b * IN + i) * FN + f]);
}

// (KW,NG) f32 -> transposed [n][k] bf16  (works for Wx and Ws)
__global__ void k_cvt_w(const float* __restrict__ src, u16* __restrict__ dst) {
  int t = blockIdx.x * blockDim.x + threadIdx.x;
  int k = t & (KW - 1);
  int n = t >> 9;
  dst[t] = f2bf(src[(size_t)k * NG + n]);
}

// C[r][n] = A[r][0:256] @ WT[n][kOff:kOff+256]^T (+ bias)
// grid = (M/64, NG/16), block = 128 (4 waves); wave w -> 16-row tile.
__global__ void k_pregemm(const u16* __restrict__ A, const u16* __restrict__ WT,
                          const float* __restrict__ bias, float* __restrict__ C,
                          int kOff) {
  int lane = threadIdx.x & 31;
  int w    = threadIdx.x >> 5;
  int r0   = blockIdx.x * 64 + w * 16;
  int n0   = blockIdx.y * 16;
  v8f acc = {};
#pragma unroll
  for (int kt = 0; kt < 8; ++kt) {
    v16bf a = load_a(A, FN, r0, kt * 32, lane);
    v16bf b = load_b(WT, KW, n0, kOff + kt * 32, lane);
    acc = __builtin_amdgcn_wmma_f32_16x16x32_bf16(false, a, false, b,
                                                  (short)0, acc, false, false);
  }
  int half = lane >> 4, col = lane & 15;
  float bv = bias ? bias[n0 + col] : 0.0f;
#pragma unroll
  for (int e = 0; e < 8; ++e)
    C[(size_t)(r0 + e + 8 * half) * NG + n0 + col] = acc[e] + bv;
}

// One diagonal step. grid = (ncells, HN/16), block = 128 (4 waves),
// dynamic LDS = 80*LDP*2 bytes. Cell i = i0 + blockIdx.x, j = d - i.
// Ws tile (80 n-rows x 512 k) staged once in LDS, shared by all 4 waves.
__global__ void k_diag(const u16* __restrict__ sprev, const float* __restrict__ cprev,
                       u16* __restrict__ snext, float* __restrict__ cnext,
                       const u16* __restrict__ WsT,
                       const float* __restrict__ Ax, const float* __restrict__ Ay,
                       float* __restrict__ out, int d, int i0) {
  extern __shared__ u16 smem[];           // 80 rows x LDP u16 = 83,200 B
  int i = i0 + blockIdx.x;
  int j = d - i;
  int lane = threadIdx.x & 31;
  int w    = threadIdx.x >> 5;
  int m0   = w * 16;
  int h0   = blockIdx.y * 16;
  bool hasH = (i > 0);   // horizontal neighbor (i-1, j)
  bool hasV = (j > 0);   // vertical   neighbor (i, j-1)

  // Warm L2 for the epilogue's Ax/Ay rows while the WMMA loop runs.
  __builtin_prefetch(Ax + (size_t)(i * BN + m0 + (lane & 15)) * NG + h0, 0, 0);
  __builtin_prefetch(Ay + (size_t)(j * BN + m0 + (lane & 15)) * NG + h0, 0, 0);

  // Cooperative stage: row r (0..79) -> n = (r/16)*256 + h0 + (r%16), k=0..511.
  // 5120 16-byte chunks / 128 threads = 40 each.
#pragma unroll 4
  for (int q = 0; q < 40; ++q) {
    int c  = threadIdx.x + (q << 7);
    int r  = c >> 6;                      // 64 chunks per row
    int rc = c & 63;
    int n  = ((r >> 4) << 8) + h0 + (r & 15);
    *(v8u*)(smem + r * LDP + rc * 8) = *(const v8u*)(WsT + (size_t)n * KW + rc * 8);
  }
  __syncthreads();

  v8f acc[5] = {{}, {}, {}, {}, {}};
#pragma unroll
  for (int kt = 0; kt < 16; ++kt) {
    bool hor = (kt < 8);
    if (hor ? !hasH : !hasV) continue;            // zero contribution
    const u16* sb = sprev + (size_t)(hor ? (i - 1) : i) * (BN * HN);
    v16bf a = load_a(sb, HN, m0, (kt & 7) * 32, lane);
    int kk = kt * 32;                              // global K into Ws (0..511)
#pragma unroll
    for (int g = 0; g < 5; ++g) {
      const u16* bp = smem + (size_t)(g * 16 + (lane & 15)) * LDP + kk + (lane >> 4) * 8;
      v16bf b = load_b_lds(bp);
      acc[g] = __builtin_amdgcn_wmma_f32_16x16x32_bf16(false, a, false, b,
                                                       (short)0, acc[g], false, false);
    }
  }

  int half = lane >> 4, col = lane & 15;
  int h = h0 + col;
#pragma unroll
  for (int e = 0; e < 8; ++e) {
    int b = m0 + e + 8 * half;
    size_t ra = (size_t)(i * BN + b) * NG + h;
    size_t rb = (size_t)(j * BN + b) * NG + h;
    float p0 = acc[0][e] + Ax[ra]          + Ay[rb];
    float p1 = acc[1][e] + Ax[ra + HN]     + Ay[rb + HN];
    float p2 = acc[2][e] + Ax[ra + 2 * HN] + Ay[rb + 2 * HN];
    float p3 = acc[3][e] + Ax[ra + 3 * HN] + Ay[rb + 3 * HN];
    float p4 = acc[4][e] + Ax[ra + 4 * HN] + Ay[rb + 4 * HN];
    float ig = 1.0f / (1.0f + __expf(-p0));
    float fg = 1.0f / (1.0f + __expf(-p1));
    float og = 1.0f / (1.0f + __expf(-p2));
    float lg = 1.0f / (1.0f + __expf(-p3));
    float gg = tanhf(p4);
    float ch = hasH ? cprev[(size_t)((i - 1) * BN + b) * HN + h] : 0.0f;
    float cv = hasV ? cprev[(size_t)(i * BN + b) * HN + h] : 0.0f;
    float c  = fg * (lg * ch + (1.0f - lg) * cv) + ig * gg;
    float s  = og * tanhf(c);
    cnext[(size_t)(i * BN + b) * HN + h] = c;
    snext[(size_t)(i * BN + b) * HN + h] = f2bf(s);
    out[(size_t)((i * ON + j) * BN + b) * HN + h] = s;
  }
}

extern "C" void kernel_launch(void* const* d_in, const int* in_sizes, int n_in,
                              void* d_out, int out_size, void* d_ws, size_t ws_size,
                              hipStream_t stream) {
  const float* x    = (const float*)d_in[0];
  const float* y    = (const float*)d_in[1];
  const float* Wx   = (const float*)d_in[2];
  const float* Ws   = (const float*)d_in[3];
  const float* bias = (const float*)d_in[4];
  float* out = (float*)d_out;

  char* ws = (char*)d_ws;
  const size_t nAct = (size_t)IN * BN * FN;   // 786432
  const size_t nW   = (size_t)NG * KW;        // 655360
  const size_t nPre = (size_t)IN * BN * NG;   // 3932160
  const size_t nSt  = (size_t)IN * BN * HN;   // 786432

  u16* xT  = (u16*)ws;
  u16* yT  = xT + nAct;
  u16* WxT = yT + nAct;
  u16* WsT = WxT + nW;
  float* Ax = (float*)(WsT + nW);
  float* Ay = Ax + nPre;
  float* c0 = Ay + nPre;
  float* c1 = c0 + nSt;
  u16* s0 = (u16*)(c1 + nSt);
  u16* s1 = s0 + nSt;

  // bf16 conversion / transposition
  k_cvt_act<<<dim3((unsigned)(nAct / 256)), 256, 0, stream>>>(x, xT);
  k_cvt_act<<<dim3((unsigned)(nAct / 256)), 256, 0, stream>>>(y, yT);
  k_cvt_w<<<dim3((unsigned)(nW / 256)), 256, 0, stream>>>(Wx, WxT);
  k_cvt_w<<<dim3((unsigned)(nW / 256)), 256, 0, stream>>>(Ws, WsT);

  // non-recurrent halves: Ax = xT @ Wx[0:256], Ay = yT @ Wx[256:512] + b
  k_pregemm<<<dim3(IN * BN / 64, NG / 16), 128, 0, stream>>>(xT, WxT, nullptr, Ax, 0);
  k_pregemm<<<dim3(ON * BN / 64, NG / 16), 128, 0, stream>>>(yT, WxT, bias, Ay, FN);

  // diagonal wavefront with ping-pong state buffers
  const size_t ldsBytes = (size_t)80 * LDP * sizeof(u16);   // 83,200 B
  u16* sp = s0; u16* sn = s1;
  float* cp = c0; float* cn = c1;
  for (int dgn = 0; dgn < IN + ON - 1; ++dgn) {
    int lo = dgn - (ON - 1); if (lo < 0) lo = 0;
    int hi = (dgn < IN - 1) ? dgn : (IN - 1);
    int nc = hi - lo + 1;
    k_diag<<<dim3(nc, HN / 16), 128, ldsBytes, stream>>>(sp, cp, sn, cn, WsT,
                                                         Ax, Ay, out, dgn, lo);
    u16* ts = sp; sp = sn; sn = ts;
    float* tc = cp; cp = cn; cn = tc;
  }
}
